// ContinuousTimeAwareMHSA_51840255263282
// MI455X (gfx1250) — matrix operand order repaired and checked by reading, not compiled, not run on hardware
//
#include <hip/hip_runtime.h>
#include <hip/hip_bf16.h>

#define HID   1024
#define HEADS 16
#define HDIM  64
#define BATCH 4
#define SEQ   2048

#define DEVI __device__ __forceinline__

typedef __attribute__((ext_vector_type(16))) __bf16 v16bf;
typedef __attribute__((ext_vector_type(8)))  float  v8f;

union Frag { uint4 q[2]; v16bf v; };

// ---- helpers -------------------------------------------------------------

DEVI unsigned short f2bf(float f) {            // RNE f32 -> bf16
    unsigned int u = __float_as_uint(f);
    u += 0x7FFFu + ((u >> 16) & 1u);
    return (unsigned short)(u >> 16);
}

// wave-local LDS ordering: all prior ds ops complete (gfx1250 split counter)
DEVI void lds_fence_wave() { asm volatile("s_wait_dscnt 0x0" ::: "memory"); }

// Async DMA global->LDS, 16B per lane, tracked by ASYNCcnt (ISA 15.18.3 op 98).
// LDS address = low 32 bits of the flat pointer.
DEVI void async_copy_b128(void* lds, const void* gsrc) {
    unsigned l = (unsigned)(size_t)lds;
    asm volatile("global_load_async_to_lds_b128 %0, %1, off"
                 :: "v"(l), "v"(gsrc) : "memory");
}
DEVI void wait_async0() { asm volatile("s_wait_asynccnt 0x0" ::: "memory"); }

// A-matrix 16x32 bf16 fragment from row-major [.][ld] LDS (ISA 7.12.2):
// lane m (0-15): V0-3 = k 0..7, V4-7 = k 16..23 ; lanes 16-31: k +8.
DEVI v16bf load_fragA(const unsigned short* base, int row0, int ld, int koff, int lane) {
    const int grp = (lane >> 4) & 1, m = lane & 15;
    const unsigned short* p = base + (size_t)(row0 + m) * ld + koff + grp * 8;
    Frag f;
    f.q[0] = *(const uint4*)(p);
    f.q[1] = *(const uint4*)(p + 16);
    return f.v;
}

// B-matrix 32x16 bf16 fragment; element (k,n) stored at base[(n0+n)*ld + koff + k]
// lane n (0-15): V0-3 = k 0..7, V4-7 = k 16..23 ; lanes 16-31: k +8 (matches A).
DEVI v16bf load_fragBT(const unsigned short* base, int n0, int ld, int koff, int lane) {
    const int grp = (lane >> 4) & 1, n = lane & 15;
    const unsigned short* p = base + (size_t)(n0 + n) * ld + koff + grp * 8;
    Frag f;
    f.q[0] = *(const uint4*)(p);
    f.q[1] = *(const uint4*)(p + 16);
    return f.v;
}

DEVI v8f wmma_bf16(v16bf a, v16bf b, v8f c) {
    return __builtin_amdgcn_wmma_f32_16x16x32_bf16(false, a, false, b,
                                                   (short)0, c, false, false);
}

// ---- kernel 1a: f32 -> bf16 conversion (x) -------------------------------

__global__ void cvt_bf16_kernel(const float* __restrict__ src,
                                unsigned short* __restrict__ dst, int n) {
    int i = blockIdx.x * 256 + threadIdx.x;
    if (i < n) dst[i] = f2bf(src[i]);
}

// ---- kernel 1b: f32 -> bf16 + transpose (W -> W^T, [n][k]) ---------------

__global__ __launch_bounds__(256) void cvt_transpose_kernel(
    const float* __restrict__ src,           // [K=HID][N=HID] row-major
    unsigned short* __restrict__ dst)        // [N=HID][K=HID] row-major
{
    __shared__ float tile[32][33];
    const int n0 = blockIdx.x * 32, k0 = blockIdx.y * 32;
    const int tx = threadIdx.x & 31;
    const int ty = (threadIdx.x >> 5) * 4;
    #pragma unroll
    for (int j = 0; j < 4; ++j)
        tile[ty + j][tx] = src[(size_t)(k0 + ty + j) * HID + n0 + tx];
    __syncthreads();
    #pragma unroll
    for (int j = 0; j < 4; ++j)
        dst[(size_t)(n0 + ty + j) * HID + k0 + tx] = f2bf(tile[tx][ty + j]);
}

// ---- kernel 2: QKV projection GEMM (bf16 WMMA, f32 accum) ----------------
// C[8192,1024] = Xb @ W + bias. W pre-transposed [n][k].
// Double-buffered LDS filled by async DMA (no VGPR staging -> no spills).
// Q,K stored [B][H][S][D]; V stored transposed [B][H][D][S] for flash PV.

__global__ __launch_bounds__(128) void qkv_proj_kernel(
    const unsigned short* __restrict__ Xb,     // [8192][1024]
    const unsigned short* __restrict__ Wtb,    // [3][N=1024][K=1024]
    const float* __restrict__ bq, const float* __restrict__ bk,
    const float* __restrict__ bv,
    unsigned short* __restrict__ Qb, unsigned short* __restrict__ Kb,
    unsigned short* __restrict__ Vtb)
{
    const int p = blockIdx.z;
    const unsigned short* Wt = Wtb + (size_t)p * HID * HID;
    const float* bias = (p == 0) ? bq : (p == 1) ? bk : bv;
    unsigned short* Out = (p == 0) ? Qb : (p == 1) ? Kb : Vtb;
    const bool vmode = (p == 2);

    __shared__ __align__(16) unsigned short lx[2][64 * 64];  // X tile row-major
    __shared__ __align__(16) unsigned short lw[2][64 * 64];  // W^T tile [n][k]

    const int tid = threadIdx.x, lane = tid & 31, wv = tid >> 5;
    const int m0 = blockIdx.x * 64, n0 = blockIdx.y * 64;

    auto fetch_async = [&](int buf, int kk) {
        #pragma unroll
        for (int it = 0; it < 4; ++it) {
            int e = tid + it * 128, r = e >> 3, c8 = (e & 7) * 8;
            async_copy_b128(&lx[buf][r * 64 + c8],
                            &Xb[(size_t)(m0 + r) * HID + kk + c8]);
            async_copy_b128(&lw[buf][r * 64 + c8],
                            &Wt[(size_t)(n0 + r) * HID + kk + c8]);
        }
    };

    v8f acc[4] = {};
    fetch_async(0, 0);
    wait_async0();
    __syncthreads();
    const int NIT = HID / 64;
    for (int i = 0; i < NIT; ++i) {
        const int buf = i & 1;
        if (i + 1 < NIT) fetch_async(buf ^ 1, (i + 1) * 64);   // DMA overlaps WMMAs
        #pragma unroll
        for (int kc = 0; kc < 2; ++kc) {
            v16bf a = load_fragA(lx[buf], wv * 16, 64, kc * 32, lane);
            #pragma unroll
            for (int t = 0; t < 4; ++t) {
                v16bf b = load_fragBT(lw[buf], t * 16, 64, kc * 32, lane);
                acc[t] = wmma_bf16(a, b, acc[t]);
            }
        }
        if (i + 1 < NIT) { wait_async0(); __syncthreads(); }
    }

    // epilogue: +bias, permute into [B][H][S][D] (Q,K) or [B][H][D][S] (V)
    const int grp = lane >> 4, nl = lane & 15;
    #pragma unroll
    for (int t = 0; t < 4; ++t) {
        int n = n0 + t * 16 + nl;
        int h = n >> 6, d = n & 63;
        float bval = bias[n];
        #pragma unroll
        for (int r = 0; r < 8; ++r) {
            int mgl = m0 + wv * 16 + r + grp * 8;     // 0..8191
            int bb = mgl >> 11, s = mgl & (SEQ - 1);
            size_t base = (size_t)bb * HEADS + h;
            size_t idx = vmode ? (base * HDIM + d) * SEQ + s
                               : (base * SEQ + s) * HDIM + d;
            Out[idx] = f2bf(acc[t][r] + bval);
        }
    }
}

// ---- kernel 3: fused flash attention with time-decay bias ----------------
// one WG per (b, h, 64-row q tile); wave owns a 16-row strip.
// K/V chunks double-buffered in LDS via async DMA; V pre-transposed [d][key].

__global__ __launch_bounds__(128) void flash_attn_kernel(
    const unsigned short* __restrict__ Qb,
    const unsigned short* __restrict__ Kb,
    const unsigned short* __restrict__ Vt,   // [B][H][D][S]
    const float* __restrict__ tint,          // [B][S][S]
    const int*   __restrict__ mask,          // [B][1][S][S]
    const float* __restrict__ alpha_p,
    float* __restrict__ out)                 // [B][S][HID]
{
    __shared__ __align__(16) unsigned short lq[64 * 64];       // Q tile
    __shared__ __align__(16) unsigned short lk[2][32 * 64];    // K [key][d]
    __shared__ __align__(16) unsigned short lvt[2][64 * 32];   // V^T [d][key]
    __shared__ __align__(16) unsigned short lp[4][16 * 32];    // per-wave P

    const int tid = threadIdx.x, lane = tid & 31, wv = tid >> 5;
    const int bh = blockIdx.y;                 // b*16 + h
    const int bb = bh >> 4, h = bh & 15;
    const int q0 = blockIdx.x * 64;
    const float nalpha = -fabsf(alpha_p[0]);
    const float scale = 0.125f;                // 1/sqrt(HDIM)

    const unsigned short* Qg = Qb + ((size_t)bh * SEQ + q0) * HDIM;
    const unsigned short* Kg = Kb + (size_t)bh * SEQ * HDIM;
    const unsigned short* Vg = Vt + (size_t)bh * HDIM * SEQ;
    const float* tg  = tint + (size_t)bb * SEQ * SEQ;
    const int*   mgp = mask + (size_t)bb * SEQ * SEQ;

    auto fetch_async = [&](int buf, int k0) {
        #pragma unroll
        for (int it = 0; it < 2; ++it) {             // K chunk 32x64
            int e = tid + it * 128, r = e >> 3, c8 = (e & 7) * 8;
            async_copy_b128(&lk[buf][r * 64 + c8],
                            &Kg[((size_t)(k0 + r)) * HDIM + c8]);
        }
        #pragma unroll
        for (int it = 0; it < 2; ++it) {             // V^T chunk 64x32
            int e = tid + it * 128, d = e >> 2, c8 = (e & 3) * 8;
            async_copy_b128(&lvt[buf][d * 32 + c8],
                            &Vg[(size_t)d * SEQ + k0 + c8]);
        }
    };

    // stage Q (async) + first K/V chunk
    #pragma unroll
    for (int it = 0; it < 4; ++it) {
        int e = tid + it * 128, r = e >> 3, c8 = (e & 7) * 8;
        async_copy_b128(&lq[r * 64 + c8], &Qg[(size_t)r * HDIM + c8]);
    }
    fetch_async(0, 0);
    wait_async0();
    __syncthreads();

    const int grp = lane >> 4, nl = lane & 15;
    // Q A-fragments are chunk-invariant: hoist out of the key loop
    v16bf aq[2] = { load_fragA(lq, wv * 16, 64, 0,  lane),
                    load_fragA(lq, wv * 16, 64, 32, lane) };

    v8f o[4] = {};
    float mrow[8], lrow[8];
    #pragma unroll
    for (int r = 0; r < 8; ++r) { mrow[r] = -3.0e38f; lrow[r] = 0.0f; }

    const int NCH = SEQ / 32;
    for (int i = 0; i < NCH; ++i) {
        const int k0 = i * 32, buf = i & 1;
        if (i + 1 < NCH) fetch_async(buf ^ 1, k0 + 32);   // DMA overlaps compute

        // scores S = Q K^T : 2 key-subtiles, contraction d=64 in 2 WMMAs
        v8f sc[2] = {};
        #pragma unroll
        for (int kc = 0; kc < 2; ++kc) {
            #pragma unroll
            for (int t = 0; t < 2; ++t) {
                v16bf b = load_fragBT(lk[buf], t * 16, 64, kc * 32, lane);
                sc[t] = wmma_bf16(aq[kc], b, sc[t]);
            }
        }

        // prefetch next chunk's bias rows (global_prefetch_b8)
        if (k0 + 32 < SEQ) {
            size_t pref = (size_t)(q0 + wv * 16 + grp * 8) * SEQ + (k0 + 32) + nl;
            __builtin_prefetch(&tg[pref], 0, 1);
            __builtin_prefetch(&mgp[pref], 0, 1);
        }

        // scale + time-decay bias + mask (C layout: lane<16 -> M=r, else M=r+8)
        float sv[2][8];
        #pragma unroll
        for (int t = 0; t < 2; ++t) {
            int key = k0 + t * 16 + nl;
            #pragma unroll
            for (int r = 0; r < 8; ++r) {
                int qrow = q0 + wv * 16 + r + grp * 8;
                size_t off = (size_t)qrow * SEQ + key;
                float s = sc[t][r] * scale + nalpha * tg[off];
                sv[t][r] = (mgp[off] == 0) ? -1.0e9f : s;
            }
        }

        // online softmax per row (16-lane xor reductions), rescale accumulator
        #pragma unroll
        for (int r = 0; r < 8; ++r) {
            float mx = fmaxf(sv[0][r], sv[1][r]);
            #pragma unroll
            for (int d = 8; d >= 1; d >>= 1)
                mx = fmaxf(mx, __shfl_xor(mx, d, 32));
            float mnew = fmaxf(mrow[r], mx);
            float corr = __expf(mrow[r] - mnew);
            mrow[r] = mnew;
            float p0 = __expf(sv[0][r] - mnew);
            float p1 = __expf(sv[1][r] - mnew);
            float ps = p0 + p1;
            #pragma unroll
            for (int d = 8; d >= 1; d >>= 1)
                ps += __shfl_xor(ps, d, 32);
            lrow[r] = lrow[r] * corr + ps;
            #pragma unroll
            for (int t = 0; t < 4; ++t) o[t][r] *= corr;
            // scoreboard P (C layout -> row-major [m][k]) for A-fragment reload
            lp[wv][(r + 8 * grp) * 32 + nl]      = f2bf(p0);
            lp[wv][(r + 8 * grp) * 32 + 16 + nl] = f2bf(p1);
        }
        lds_fence_wave();   // lp is wave-private: s_wait_dscnt 0 suffices

        // O += P(16x32) @ V(32x64)
        v16bf a = load_fragA(lp[wv], 0, 32, 0, lane);
        #pragma unroll
        for (int t = 0; t < 4; ++t) {
            v16bf b = load_fragBT(lvt[buf], t * 16, 32, 0, lane);
            o[t] = wmma_bf16(a, b, o[t]);
        }

        if (i + 1 < NCH) { wait_async0(); __syncthreads(); }
    }

    // epilogue: normalize and scatter to [B][S][HID]
    #pragma unroll
    for (int r = 0; r < 8; ++r) {
        float inv = 1.0f / lrow[r];
        int qrow = q0 + wv * 16 + r + grp * 8;
        float* op = out + ((size_t)bb * SEQ + qrow) * HID + h * HDIM;
        #pragma unroll
        for (int t = 0; t < 4; ++t)
            op[t * 16 + nl] = o[t][r] * inv;
    }
}

// ---- launcher ------------------------------------------------------------

extern "C" void kernel_launch(void* const* d_in, const int* in_sizes, int n_in,
                              void* d_out, int out_size, void* d_ws, size_t ws_size,
                              hipStream_t stream) {
    (void)in_sizes; (void)n_in; (void)out_size; (void)ws_size;
    const float* x    = (const float*)d_in[0];
    const float* tint = (const float*)d_in[1];
    const int*   mask = (const int*)d_in[2];
    const float* Wq   = (const float*)d_in[3];
    const float* bq   = (const float*)d_in[4];
    const float* Wk   = (const float*)d_in[5];
    const float* bk   = (const float*)d_in[6];
    const float* Wv   = (const float*)d_in[7];
    const float* bv   = (const float*)d_in[8];
    const float* alpha= (const float*)d_in[9];
    float* out = (float*)d_out;

    const size_t NX = (size_t)BATCH * SEQ * HID;      // 8,388,608
    const size_t NW = (size_t)HID * HID;              // 1,048,576
    char* ws = (char*)d_ws;
    unsigned short* Xb  = (unsigned short*)ws;                      ws += NX * 2;
    unsigned short* Wtb = (unsigned short*)ws;                      ws += 3 * NW * 2;
    unsigned short* Qb  = (unsigned short*)ws;                      ws += NX * 2;
    unsigned short* Kb  = (unsigned short*)ws;                      ws += NX * 2;
    unsigned short* Vtb = (unsigned short*)ws;

    cvt_bf16_kernel<<<(int)(NX / 256), 256, 0, stream>>>(x, Xb, (int)NX);
    dim3 tg(HID / 32, HID / 32);
    cvt_transpose_kernel<<<tg, 256, 0, stream>>>(Wq, Wtb);
    cvt_transpose_kernel<<<tg, 256, 0, stream>>>(Wk, Wtb + NW);
    cvt_transpose_kernel<<<tg, 256, 0, stream>>>(Wv, Wtb + 2 * NW);

    qkv_proj_kernel<<<dim3(BATCH * SEQ / 64, HID / 64, 3), 128, 0, stream>>>(
        Xb, Wtb, bq, bk, bv, Qb, Kb, Vtb);

    flash_attn_kernel<<<dim3(SEQ / 64, BATCH * HEADS), 128, 0, stream>>>(
        Qb, Kb, Vtb, tint, mask, alpha, out);
}
